// CausalSelfAttention_15444702396434
// MI455X (gfx1250) — matrix operand order repaired and checked by reading, not compiled, not run on hardware
//
#include <hip/hip_runtime.h>
#include <math.h>

// ---------------- problem constants ----------------
#define SEQ   4096
#define DIM   1024
#define NHEAD 16
#define HEADD 64
#define DQKV  3072   // 3*DIM

typedef __bf16 bf16;
typedef __attribute__((ext_vector_type(16))) __bf16 bf16x16;
typedef __attribute__((ext_vector_type(8)))  __bf16 bf16x8;
typedef __attribute__((ext_vector_type(8)))  float  f32x8;
typedef __attribute__((ext_vector_type(4)))  float  f32x4;

// ---------------------------------------------------------------------------
// WMMA helper: D = A(16x32 bf16) * B(32x16 bf16) + C(16x16 f32)
// ---------------------------------------------------------------------------
__device__ __forceinline__ f32x8 wmma_bf16(bf16x16 a, bf16x16 b, f32x8 c) {
    return __builtin_amdgcn_wmma_f32_16x16x32_bf16(
        /*neg_a=*/false, a, /*neg_b=*/false, b,
        /*c_mod=*/(short)0, c, /*reuse_a=*/false, /*reuse_b=*/false);
}

// A fragment 16(M) x 32(K) from a row-major bf16 matrix, tile origin p0, row
// stride ld (elements).  ISA layout: lane m (0..15) holds row m; VGPR0..3 hold
// K = kb..kb+7, VGPR4..7 hold K = kb+16..kb+23 with kb = (lane>=16) ? 8 : 0.
__device__ __forceinline__ bf16x16 load_a16x32(const bf16* __restrict__ p0,
                                               int ld, int lane) {
    int m  = lane & 15;
    int kb = (lane >> 4) << 3;          // 0 or 8
    const bf16* p = p0 + (size_t)m * ld + kb;
    union { bf16x16 v; bf16x8 h[2]; } u;
    u.h[0] = *(const bf16x8*)(p);        // K = kb .. kb+7
    u.h[1] = *(const bf16x8*)(p + 16);   // K = kb+16 .. kb+23
    return u.v;
}

// B fragment 32(K) x 16(N) where B(k,n) = Mat[n0+n][k0+k] for a row-major
// matrix (i.e. computing X * Mat^T).  p0 = &Mat[n0][k0], ld = row stride.
// ISA layout: lane n holds column n; VGPR0..3 hold K = kb..kb+7, VGPR4..7
// hold K = kb+16..kb+23 with kb = (lane>=16) ? 8 : 0 (matches A fragment).
__device__ __forceinline__ bf16x16 load_b32x16(const bf16* __restrict__ p0,
                                               int ld, int lane) {
    int n  = lane & 15;
    int kb = (lane >> 4) << 3;          // 0 or 8
    const bf16* p = p0 + (size_t)n * ld + kb;
    union { bf16x16 v; bf16x8 h[2]; } u;
    u.h[0] = *(const bf16x8*)(p);        // K = kb .. kb+7
    u.h[1] = *(const bf16x8*)(p + 16);   // K = kb+16 .. kb+23
    return u.v;
}

// ---------------------------------------------------------------------------
// Kernel 0: fp32 -> bf16 conversion (vectorized x4)
// ---------------------------------------------------------------------------
__global__ __launch_bounds__(256)
void cvt_f32_to_bf16(const float* __restrict__ in, bf16* __restrict__ out, int n) {
    int i = (blockIdx.x * blockDim.x + threadIdx.x) * 4;
    if (i < n) {
        f32x4 v = *(const f32x4*)(in + i);
        union { bf16 b[4]; unsigned long long u; } o;
        o.b[0] = (bf16)v[0]; o.b[1] = (bf16)v[1];
        o.b[2] = (bf16)v[2]; o.b[3] = (bf16)v[3];
        *(unsigned long long*)(out + i) = o.u;
    }
}

// ---------------------------------------------------------------------------
// Kernel 1: QKV = x @ Wqkv^T + b, scattered into attention layouts.
//   Qo [h][SEQ][64]  (pre-scaled by 1/sqrt(64))
//   Ko [h][SEQ][64]
//   Vt [h][64][SEQ]  (transposed so PV B-operand loads are contiguous)
// Workgroup: 256 threads = 8 waves; tile 128(M) x 128(N); wave = 32x64.
// ---------------------------------------------------------------------------
__global__ __launch_bounds__(256)
void qkv_gemm(const bf16* __restrict__ X, const bf16* __restrict__ W,
              const float* __restrict__ bias,
              bf16* __restrict__ Qo, bf16* __restrict__ Ko, bf16* __restrict__ Vt) {
    int lane = threadIdx.x & 31;
    int w    = threadIdx.x >> 5;
    int m0   = blockIdx.x * 128 + (w & 3) * 32;
    int n0   = blockIdx.y * 128 + (w >> 2) * 64;

    f32x8 acc[2][4] = {};
    for (int k0 = 0; k0 < DIM; k0 += 32) {
        bf16x16 a0 = load_a16x32(X + (size_t)m0        * DIM + k0, DIM, lane);
        bf16x16 a1 = load_a16x32(X + (size_t)(m0 + 16) * DIM + k0, DIM, lane);
#pragma unroll
        for (int j = 0; j < 4; ++j) {
            bf16x16 b = load_b32x16(W + (size_t)(n0 + j * 16) * DIM + k0, DIM, lane);
            acc[0][j] = wmma_bf16(a0, b, acc[0][j]);
            acc[1][j] = wmma_bf16(a1, b, acc[1][j]);
        }
    }

    int hi = lane >> 4;
    int nn = lane & 15;
#pragma unroll
    for (int i = 0; i < 2; ++i) {
#pragma unroll
        for (int j = 0; j < 4; ++j) {
            int f    = n0 + j * 16 + nn;      // feature in [0, 3072)
            float bv = bias[f];
            int sec  = f >> 10;               // 0=Q 1=K 2=V
            int fq   = f & (DIM - 1);
            int head = fq >> 6;
            int d    = fq & 63;
#pragma unroll
            for (int r = 0; r < 8; ++r) {
                int t   = m0 + i * 16 + r + hi * 8;   // token index
                float v = acc[i][j][r] + bv;
                if (sec == 0) {
                    Qo[((size_t)head * SEQ + t) * HEADD + d] = (bf16)(v * 0.125f);
                } else if (sec == 1) {
                    Ko[((size_t)head * SEQ + t) * HEADD + d] = (bf16)v;
                } else {
                    Vt[((size_t)head * HEADD + d) * SEQ + t] = (bf16)v;
                }
            }
        }
    }
}

// ---------------------------------------------------------------------------
// Kernel 2: causal flash attention per head.
// Workgroup: 256 threads = 8 waves; 128 query rows per block (16 per wave).
// Wave loop over 32-key blocks: 4 WMMAs for S = Q*K^T, online softmax with
// 16-lane xor-shuffle reductions, P transposed C->A layout through LDS,
// 4 WMMAs for O += P*V.  Output Ob [SEQ][DIM] bf16 (heads concatenated).
// ---------------------------------------------------------------------------
__global__ __launch_bounds__(256)
void attn_fwd(const bf16* __restrict__ Q, const bf16* __restrict__ K,
              const bf16* __restrict__ Vt, bf16* __restrict__ Ob) {
    __shared__ bf16 lds_p[8 * 16 * 32];   // 1KB per wave P tile (16x32)

    int lane = threadIdx.x & 31;
    int w    = threadIdx.x >> 5;
    int h    = blockIdx.y;
    int q0   = blockIdx.x * 128 + w * 16;

    const bf16* Qh = Q  + (size_t)h * SEQ * HEADD;
    const bf16* Kh = K  + (size_t)h * SEQ * HEADD;
    const bf16* Vh = Vt + (size_t)h * HEADD * SEQ;
    bf16* myP = lds_p + w * (16 * 32);

    // Q fragments for this wave's 16 rows (already scaled by 1/sqrt(d))
    bf16x16 aq0 = load_a16x32(Qh + (size_t)q0 * HEADD + 0,  HEADD, lane);
    bf16x16 aq1 = load_a16x32(Qh + (size_t)q0 * HEADD + 32, HEADD, lane);

    int hi = lane >> 4;
    int nn = lane & 15;

    float mrow[8], lrow[8];
    f32x8 acc[4] = {};
#pragma unroll
    for (int r = 0; r < 8; ++r) { mrow[r] = -__builtin_inff(); lrow[r] = 0.f; }

    int nblocks = (q0 + 47) >> 5;   // keys 0 .. q0+15 inclusive
    for (int kb = 0; kb < nblocks; ++kb) {
        int ks = kb * 32;

        // ---- S = Q * K^T  (16 q x 32 k) ----
        f32x8 s[2];
#pragma unroll
        for (int g = 0; g < 2; ++g) {
            f32x8 z = {};
            bf16x16 bk0 = load_b32x16(Kh + (size_t)(ks + g * 16) * HEADD + 0,  HEADD, lane);
            z = wmma_bf16(aq0, bk0, z);
            bf16x16 bk1 = load_b32x16(Kh + (size_t)(ks + g * 16) * HEADD + 32, HEADD, lane);
            z = wmma_bf16(aq1, bk1, z);
            s[g] = z;
        }

        // ---- causal mask (only diagonal-straddling blocks) ----
        if (ks + 31 > q0) {
#pragma unroll
            for (int g = 0; g < 2; ++g) {
                int key = ks + g * 16 + nn;
#pragma unroll
                for (int r = 0; r < 8; ++r) {
                    int qi = q0 + r + hi * 8;
                    if (key > qi) s[g][r] = -__builtin_inff();
                }
            }
        }

        // ---- online softmax (row lives across 16 lanes of a half-wave) ----
#pragma unroll
        for (int r = 0; r < 8; ++r) {
            float rmax = fmaxf(s[0][r], s[1][r]);
#pragma unroll
            for (int off = 8; off; off >>= 1)
                rmax = fmaxf(rmax, __shfl_xor(rmax, off, 32));
            float mnew  = fmaxf(mrow[r], rmax);
            float alpha = __expf(mrow[r] - mnew);
            float p0    = __expf(s[0][r] - mnew);
            float p1    = __expf(s[1][r] - mnew);
            float rsum  = p0 + p1;
#pragma unroll
            for (int off = 8; off; off >>= 1)
                rsum += __shfl_xor(rsum, off, 32);
            mrow[r] = mnew;
            lrow[r] = lrow[r] * alpha + rsum;
#pragma unroll
            for (int j = 0; j < 4; ++j) acc[j][r] *= alpha;

            int row = r + hi * 8;
            myP[row * 32 + nn]      = (bf16)p0;
            myP[row * 32 + 16 + nn] = (bf16)p1;
        }

        // ---- transpose P from C-layout to A-layout through LDS ----
        asm volatile("s_wait_dscnt 0x0" ::: "memory");
        bf16x16 ap;
        {
            int m  = lane & 15;
            int kbo = (lane >> 4) << 3;
            union { bf16x16 v; bf16x8 t[2]; } u;
            u.t[0] = *(const bf16x8*)(&myP[m * 32 + kbo]);
            u.t[1] = *(const bf16x8*)(&myP[m * 32 + kbo + 16]);
            ap = u.v;
        }
        asm volatile("" ::: "memory");

        // ---- O += P * V  (Vt rows are head-dim, contiguous keys) ----
#pragma unroll
        for (int j = 0; j < 4; ++j) {
            bf16x16 bv = load_b32x16(Vh + (size_t)(j * 16) * SEQ + ks, SEQ, lane);
            acc[j] = wmma_bf16(ap, bv, acc[j]);
        }
    }

    // ---- normalize and store (bf16, heads concatenated along DIM) ----
#pragma unroll
    for (int r = 0; r < 8; ++r) {
        float inv = 1.0f / lrow[r];
        int t = q0 + r + hi * 8;
#pragma unroll
        for (int j = 0; j < 4; ++j) {
            Ob[(size_t)t * DIM + h * HEADD + j * 16 + nn] = (bf16)(acc[j][r] * inv);
        }
    }
}

// ---------------------------------------------------------------------------
// Kernel 3: out = Ob @ Wout^T + b_out  -> fp32 d_out.
// Same tiling as kernel 1.
// ---------------------------------------------------------------------------
__global__ __launch_bounds__(256)
void out_gemm(const bf16* __restrict__ A, const bf16* __restrict__ W,
              const float* __restrict__ bias, float* __restrict__ C) {
    int lane = threadIdx.x & 31;
    int w    = threadIdx.x >> 5;
    int m0   = blockIdx.x * 128 + (w & 3) * 32;
    int n0   = blockIdx.y * 128 + (w >> 2) * 64;

    f32x8 acc[2][4] = {};
    for (int k0 = 0; k0 < DIM; k0 += 32) {
        bf16x16 a0 = load_a16x32(A + (size_t)m0        * DIM + k0, DIM, lane);
        bf16x16 a1 = load_a16x32(A + (size_t)(m0 + 16) * DIM + k0, DIM, lane);
#pragma unroll
        for (int j = 0; j < 4; ++j) {
            bf16x16 b = load_b32x16(W + (size_t)(n0 + j * 16) * DIM + k0, DIM, lane);
            acc[0][j] = wmma_bf16(a0, b, acc[0][j]);
            acc[1][j] = wmma_bf16(a1, b, acc[1][j]);
        }
    }

    int hi = lane >> 4;
    int nn = lane & 15;
#pragma unroll
    for (int i = 0; i < 2; ++i) {
#pragma unroll
        for (int j = 0; j < 4; ++j) {
            int f    = n0 + j * 16 + nn;
            float bv = bias[f];
#pragma unroll
            for (int r = 0; r < 8; ++r) {
                int t = m0 + i * 16 + r + hi * 8;
                C[(size_t)t * DIM + f] = acc[i][j][r] + bv;
            }
        }
    }
}

// ---------------------------------------------------------------------------
// Host launcher
// ---------------------------------------------------------------------------
extern "C" void kernel_launch(void* const* d_in, const int* in_sizes, int n_in,
                              void* d_out, int out_size, void* d_ws, size_t ws_size,
                              hipStream_t stream) {
    const float* x    = (const float*)d_in[0];   // [SEQ, DIM]
    const float* Wqkv = (const float*)d_in[1];   // [3*DIM, DIM]
    const float* bqkv = (const float*)d_in[2];   // [3*DIM]
    const float* Wout = (const float*)d_in[3];   // [DIM, DIM]
    const float* bout = (const float*)d_in[4];   // [DIM]
    float* out = (float*)d_out;                  // [SEQ, DIM] fp32

    // workspace layout (all bf16)
    char* ws = (char*)d_ws;
    bf16* xb  = (bf16*)ws;  ws += (size_t)SEQ  * DIM * sizeof(bf16);
    bf16* wqb = (bf16*)ws;  ws += (size_t)DQKV * DIM * sizeof(bf16);
    bf16* wob = (bf16*)ws;  ws += (size_t)DIM  * DIM * sizeof(bf16);
    bf16* Qb  = (bf16*)ws;  ws += (size_t)NHEAD * SEQ * HEADD * sizeof(bf16);
    bf16* Kb  = (bf16*)ws;  ws += (size_t)NHEAD * SEQ * HEADD * sizeof(bf16);
    bf16* Vt  = (bf16*)ws;  ws += (size_t)NHEAD * HEADD * SEQ * sizeof(bf16);
    bf16* Ob  = (bf16*)ws;  ws += (size_t)SEQ * DIM * sizeof(bf16);

    // 0) fp32 -> bf16 conversions
    cvt_f32_to_bf16<<<(SEQ * DIM / 4 + 255) / 256, 256, 0, stream>>>(x, xb, SEQ * DIM);
    cvt_f32_to_bf16<<<(DQKV * DIM / 4 + 255) / 256, 256, 0, stream>>>(Wqkv, wqb, DQKV * DIM);
    cvt_f32_to_bf16<<<(DIM * DIM / 4 + 255) / 256, 256, 0, stream>>>(Wout, wob, DIM * DIM);

    // 1) QKV projection + layout scatter
    dim3 g1(SEQ / 128, DQKV / 128);
    qkv_gemm<<<g1, 256, 0, stream>>>(xb, wqb, bqkv, Qb, Kb, Vt);

    // 2) causal attention
    dim3 g2(SEQ / 128, NHEAD);
    attn_fwd<<<g2, 256, 0, stream>>>(Qb, Kb, Vt, Ob);

    // 3) output projection (fp32 out + bias)
    dim3 g3(SEQ / 128, DIM / 128);
    out_gemm<<<g3, 256, 0, stream>>>(Ob, wob, bout, out);
}
